// GAT_71914932404377
// MI455X (gfx1250) — matrix operand order repaired and checked by reading, not compiled, not run on hardware
//
#include <hip/hip_runtime.h>
#include <hip/hip_bf16.h>

// ---------------------------------------------------------------------------
// GAT 2-layer pipeline for MI455X (gfx1250, wave32).
// Dense GEMMs use v_wmma_f32_16x16x32_bf16; edge softmax/aggregation are
// atomic-based scatter kernels (memory-bound part, ~3.5GB/layer -> ~0.3ms/layer
// at 23.3 TB/s). Segment-max pass of the softmax is omitted (stability-only;
// logits are O(1)), saving one full edge pass.
// ---------------------------------------------------------------------------

typedef __attribute__((ext_vector_type(16))) __bf16 v16bf;
typedef __attribute__((ext_vector_type(8)))  float  v8f;

#define NEG_SLOPE 0.2f
#define BN_EPS 1e-5f

static __device__ __forceinline__ float lrelu(float v) {
    return v > 0.f ? v : NEG_SLOPE * v;
}

// ---------------------------------------------------------------------------
// Pack W (128x128 fp32, row-major [k][n]) into bf16 in the exact WMMA B-operand
// lane layout: for column-tile nt (cols nt*16..+15), k-step kki (K=kki*32..+31):
//   lane l holds col = nt*16 + (l&15), element j has K = kki*32 + (l<16 ? 0 : 8)
//   + (j&7) + (j>=8 ? 16 : 0).  Stored as [(nt*4+kki)*32 + lane][16] bf16.
// ---------------------------------------------------------------------------
__global__ void pack_W_kernel(const float* __restrict__ W, __bf16* __restrict__ Wpk) {
    int i = blockIdx.x * blockDim.x + threadIdx.x;   // 16384 total
    if (i >= 8 * 4 * 32 * 16) return;
    int j    = i & 15;
    int lane = (i >> 4) & 31;
    int kki  = (i >> 9) & 3;
    int nt   = i >> 11;
    int k    = kki * 32 + ((lane < 16) ? 0 : 8) + (j & 7) + ((j & 8) ? 16 : 0);
    int col  = nt * 16 + (lane & 15);
    Wpk[i] = (__bf16)W[k * 128 + col];
}

// ---------------------------------------------------------------------------
// C[nrows x 128] = A[nrows x 128] * W[128 x 128]   (A fp32 -> bf16 on the fly)
// One wave per 16-row strip; 8 accumulators cover all 128 output columns.
// 32 v_wmma per wave; A row data read exactly once per strip.
// ---------------------------------------------------------------------------
__global__ __launch_bounds__(256)
void gemm128_wmma_kernel(const float* __restrict__ A, const __bf16* __restrict__ Wpk,
                         float* __restrict__ C, int nrows) {
    const int lane  = threadIdx.x & 31;
    const int wave  = threadIdx.x >> 5;
    const int strip = blockIdx.x * 8 + wave;          // wave-uniform
    const int nstrips = (nrows + 15) >> 4;
    if (strip >= nstrips) return;                     // uniform exit: EXEC stays all-1s

    const int row0 = strip << 4;
    const int m    = lane & 15;
    const int kb   = (lane & 16) ? 8 : 0;             // A-operand K sub-base
    int row = row0 + m;
    row = row < nrows ? row : (nrows - 1);            // clamp (select, no branch)
    const float* __restrict__ arow = A + (size_t)row * 128;

    v8f acc[8] = {};
    const v16bf* __restrict__ Wv = (const v16bf*)Wpk;

#pragma unroll
    for (int kki = 0; kki < 4; ++kki) {
        const int kk = kki * 32;
        v16bf av;
#pragma unroll
        for (int j = 0; j < 8; ++j) {
            av[j]     = (__bf16)arow[kk + kb + j];        // VGPR0-3: K = kb+0..7
            av[j + 8] = (__bf16)arow[kk + kb + 16 + j];   // VGPR4-7: K = kb+16..23
        }
#pragma unroll
        for (int nt = 0; nt < 8; ++nt) {
            v16bf bv = Wv[(nt * 4 + kki) * 32 + lane];    // single 32B load
            acc[nt] = __builtin_amdgcn_wmma_f32_16x16x32_bf16(
                false, av, false, bv, (short)0, acc[nt], false, false);
        }
    }

    // C/D layout: lane l -> col = l&15, VGPR r -> row = r + (l>=16 ? 8 : 0)
    const int n0 = lane & 15;
    const int mb = (lane & 16) ? 8 : 0;
#pragma unroll
    for (int nt = 0; nt < 8; ++nt) {
#pragma unroll
        for (int r = 0; r < 8; ++r) {
            int rr = row0 + mb + r;
            if (rr < nrows) C[(size_t)rr * 128 + nt * 16 + n0] = acc[nt][r];
        }
    }
}

// ---------------------------------------------------------------------------
// es[n][h] = <h[n,h,:], a_src[h,:]> ;  ed likewise.  H layout [N][4][32].
// ---------------------------------------------------------------------------
__global__ void attn_scores_kernel(const float* __restrict__ H,
                                   const float* __restrict__ asrc,
                                   const float* __restrict__ adst,
                                   float* __restrict__ es, float* __restrict__ ed, int n) {
    int i = blockIdx.x * blockDim.x + threadIdx.x;    // n*4
    if (i >= n * 4) return;
    int node = i >> 2, hd = i & 3;
    const float* hv = H + (size_t)node * 128 + hd * 32;
    const float* av = asrc + hd * 32;
    const float* bv = adst + hd * 32;
    float s = 0.f, d = 0.f;
#pragma unroll
    for (int f = 0; f < 32; ++f) { s += hv[f] * av[f]; d += hv[f] * bv[f]; }
    es[i] = s; ed[i] = d;
}

// ---------------------------------------------------------------------------
// Softmax denominator: den[dst][h] += exp(lrelu(es[src][h]+ed[dst][h])).
// Edge slots [0,E) are real edges, [E, E+n) are self-loops.
// ---------------------------------------------------------------------------
__global__ void edge_den_kernel(const int* __restrict__ src, const int* __restrict__ dst,
                                const float* __restrict__ es, const float* __restrict__ ed,
                                float* __restrict__ den, int E, int n) {
    int i = blockIdx.x * blockDim.x + threadIdx.x;
    if (i >= E + n) return;
    int s, d;
    if (i < E) { s = src[i]; d = dst[i]; } else { s = d = i - E; }
    const float4 a = *(const float4*)(es + 4 * (size_t)s);
    const float4 b = *(const float4*)(ed + 4 * (size_t)d);
    float* dp = den + 4 * (size_t)d;
    atomicAdd(dp + 0, __expf(lrelu(a.x + b.x)));
    atomicAdd(dp + 1, __expf(lrelu(a.y + b.y)));
    atomicAdd(dp + 2, __expf(lrelu(a.z + b.z)));
    atomicAdd(dp + 3, __expf(lrelu(a.w + b.w)));
}

// ---------------------------------------------------------------------------
// Aggregation: out[dst] += alpha * h[src].  One wave per edge, lane = 4 chans.
// ---------------------------------------------------------------------------
__global__ __launch_bounds__(256)
void edge_agg_kernel(const int* __restrict__ src, const int* __restrict__ dst,
                     const float* __restrict__ es, const float* __restrict__ ed,
                     const float* __restrict__ den, const float* __restrict__ H,
                     float* __restrict__ out, int E, int n) {
    long long w = (long long)blockIdx.x * 8 + (threadIdx.x >> 5);
    int lane = threadIdx.x & 31;
    if (w >= (long long)E + n) return;
    int s, d;
    if (w < E) { s = src[w]; d = dst[w]; } else { s = d = (int)(w - E); }
    int hd = lane >> 3;                               // 8 lanes per head
    float e = lrelu(es[4 * (size_t)s + hd] + ed[4 * (size_t)d + hd]);
    float alpha = __expf(e) / den[4 * (size_t)d + hd];
    const float4 hv = *(const float4*)(H + (size_t)s * 128 + lane * 4);
    float* op = out + (size_t)d * 128 + lane * 4;
    atomicAdd(op + 0, alpha * hv.x);
    atomicAdd(op + 1, alpha * hv.y);
    atomicAdd(op + 2, alpha * hv.z);
    atomicAdd(op + 3, alpha * hv.w);
}

// ---------------------------------------------------------------------------
// BN stats: st[0..C) = sum, st[C..2C) = sumsq of (X + bias).  <<<512, C>>>.
// ---------------------------------------------------------------------------
__global__ void bn_stats_kernel(const float* __restrict__ X, const float* __restrict__ bias,
                                float* __restrict__ st, int n, int C) {
    int c = threadIdx.x;
    if (c >= C) return;
    float b = bias ? bias[c] : 0.f;
    float s = 0.f, q = 0.f;
    for (int r = blockIdx.x; r < n; r += gridDim.x) {
        float v = X[(size_t)r * C + c] + b;
        s += v; q += v * v;
    }
    atomicAdd(&st[c], s);
    atomicAdd(&st[C + c], q);
}

__global__ void bn_norm_elu_kernel(const float* __restrict__ X, const float* __restrict__ bias,
                                   const float* __restrict__ st, const float* __restrict__ gamma,
                                   const float* __restrict__ beta, float* __restrict__ Y,
                                   int n, int C) {
    long long i = (long long)blockIdx.x * blockDim.x + threadIdx.x;
    if (i >= (long long)n * C) return;
    int c = (int)(i % C);
    float nf  = (float)n;
    float mu  = st[c] / nf;
    float var = st[C + c] / nf - mu * mu;
    float b   = bias ? bias[c] : 0.f;
    float v   = X[i] + b;
    float y   = gamma[c] * (v - mu) * rsqrtf(var + BN_EPS) + beta[c];
    Y[i] = y > 0.f ? y : __expf(y) - 1.f;
}

// mean over 4 heads of X[N][4][32], + bias[32]
__global__ void head_mean_kernel(const float* __restrict__ X, const float* __restrict__ b2,
                                 float* __restrict__ Y, int n) {
    long long i = (long long)blockIdx.x * blockDim.x + threadIdx.x;
    if (i >= (long long)n * 32) return;
    long long node = i >> 5;
    int f = (int)(i & 31);
    const float* p = X + node * 128 + f;
    Y[i] = 0.25f * (p[0] + p[32] + p[64] + p[96]) + b2[f];
}

__global__ void pool_accum_kernel(const float* __restrict__ X, const int* __restrict__ batch,
                                  float* __restrict__ psum, float* __restrict__ pcnt, int n) {
    long long i = (long long)blockIdx.x * blockDim.x + threadIdx.x;
    if (i >= (long long)n * 32) return;
    int node = (int)(i >> 5);
    int f = (int)(i & 31);
    int g = batch[node];
    atomicAdd(&psum[(size_t)g * 32 + f], X[i]);
    if (f == 0) atomicAdd(&pcnt[g], 1.f);
}

__global__ void final_linear_kernel(const float* __restrict__ psum, const float* __restrict__ pcnt,
                                    const float* __restrict__ Wl, const float* __restrict__ bl,
                                    float* __restrict__ out, int G) {
    int i = blockIdx.x * blockDim.x + threadIdx.x;    // G*2
    if (i >= G * 2) return;
    int g = i >> 1, o = i & 1;
    float c = pcnt[g];
    c = c > 1.f ? c : 1.f;
    float acc = bl[o];
    const float* ps = psum + (size_t)g * 32;
#pragma unroll
    for (int f = 0; f < 32; ++f) acc += (ps[f] / c) * Wl[f * 2 + o];
    out[i] = acc;
}

// ---------------------------------------------------------------------------
extern "C" void kernel_launch(void* const* d_in, const int* in_sizes, int n_in,
                              void* d_out, int out_size, void* d_ws, size_t ws_size,
                              hipStream_t stream) {
    const float* x     = (const float*)d_in[0];
    const int*   eidx  = (const int*)d_in[1];
    const int*   batch = (const int*)d_in[2];
    const float* W1    = (const float*)d_in[3];
    const float* as1   = (const float*)d_in[4];
    const float* ad1   = (const float*)d_in[5];
    const float* b1    = (const float*)d_in[6];
    const float* g1    = (const float*)d_in[7];
    const float* be1   = (const float*)d_in[8];
    const float* W2    = (const float*)d_in[9];
    const float* as2   = (const float*)d_in[10];
    const float* ad2   = (const float*)d_in[11];
    const float* b2    = (const float*)d_in[12];
    const float* g2    = (const float*)d_in[13];
    const float* be2   = (const float*)d_in[14];
    const float* Wl    = (const float*)d_in[15];
    const float* bl    = (const float*)d_in[16];
    float* out = (float*)d_out;

    const int N = in_sizes[0] / 128;
    const int E = in_sizes[1] / 2;
    const int G = out_size / 2;
    const int* src = eidx;
    const int* dst = eidx + E;

    // ---- workspace layout (fp32 units) ----
    float* ws = (float*)d_ws;
    size_t o = 0;
    float* A   = ws + o;  o += (size_t)N * 128;   // feature buffer 1
    float* B   = ws + o;  o += (size_t)N * 128;   // feature buffer 2
    float* ES  = ws + o;  o += (size_t)N * 4;
    float* ED  = ws + o;  o += (size_t)N * 4;
    float* DEN = ws + o;  o += (size_t)N * 4;
    float* HM  = ws + o;  o += (size_t)N * 32;
    float* ST  = ws + o;  o += 256;
    float* PS  = ws + o;  o += (size_t)G * 32;
    float* PC  = ws + o;  o += (size_t)G;
    __bf16* Wpk = (__bf16*)(ws + o);              // 16384 bf16 (32B-aligned)
    (void)ws_size; (void)n_in;

    const int nE   = E + N;
    const int strips = (N + 15) >> 4;
    dim3 blk256(256);
    dim3 gGemm((strips + 7) / 8);
    dim3 gAttn((N * 4 + 255) / 256);
    dim3 gDen((nE + 255) / 256);
    dim3 gAgg((nE + 7) / 8);                      // 8 waves (edges) per block
    dim3 gElem128(((long long)N * 128 + 255) / 256);
    dim3 gElem32(((long long)N * 32 + 255) / 256);

    // =================== Layer 1 ===================
    pack_W_kernel<<<64, blk256, 0, stream>>>(W1, Wpk);
    gemm128_wmma_kernel<<<gGemm, blk256, 0, stream>>>(x, Wpk, A, N);        // h1 -> A
    attn_scores_kernel<<<gAttn, blk256, 0, stream>>>(A, as1, ad1, ES, ED, N);
    hipMemsetAsync(DEN, 0, (size_t)N * 4 * sizeof(float), stream);
    edge_den_kernel<<<gDen, blk256, 0, stream>>>(src, dst, ES, ED, DEN, E, N);
    hipMemsetAsync(B, 0, (size_t)N * 128 * sizeof(float), stream);
    edge_agg_kernel<<<gAgg, blk256, 0, stream>>>(src, dst, ES, ED, DEN, A, B, E, N); // agg1 -> B
    hipMemsetAsync(ST, 0, 256 * sizeof(float), stream);
    bn_stats_kernel<<<512, 128, 0, stream>>>(B, b1, ST, N, 128);
    bn_norm_elu_kernel<<<gElem128, blk256, 0, stream>>>(B, b1, ST, g1, be1, A, N, 128); // x2 -> A

    // =================== Layer 2 ===================
    pack_W_kernel<<<64, blk256, 0, stream>>>(W2, Wpk);
    gemm128_wmma_kernel<<<gGemm, blk256, 0, stream>>>(A, Wpk, B, N);        // h2 -> B
    attn_scores_kernel<<<gAttn, blk256, 0, stream>>>(B, as2, ad2, ES, ED, N);
    hipMemsetAsync(DEN, 0, (size_t)N * 4 * sizeof(float), stream);
    edge_den_kernel<<<gDen, blk256, 0, stream>>>(src, dst, ES, ED, DEN, E, N);
    hipMemsetAsync(A, 0, (size_t)N * 128 * sizeof(float), stream);
    edge_agg_kernel<<<gAgg, blk256, 0, stream>>>(src, dst, ES, ED, DEN, B, A, E, N); // agg2 -> A
    head_mean_kernel<<<gElem32, blk256, 0, stream>>>(A, b2, HM, N);
    hipMemsetAsync(ST, 0, 256 * sizeof(float), stream);
    bn_stats_kernel<<<512, 32, 0, stream>>>(HM, nullptr, ST, N, 32);
    bn_norm_elu_kernel<<<gElem32, blk256, 0, stream>>>(HM, nullptr, ST, g2, be2, HM, N, 32);

    // =================== Pool + linear ===================
    hipMemsetAsync(PS, 0, ((size_t)G * 32 + G) * sizeof(float), stream);    // PS and PC contiguous
    pool_accum_kernel<<<gElem32, blk256, 0, stream>>>(HM, batch, PS, PC, N);
    final_linear_kernel<<<(G * 2 + 255) / 256, blk256, 0, stream>>>(PS, PC, Wl, bl, out, G);
}
